// MaskedLinear_24232205484135
// MI455X (gfx1250) — matrix operand order repaired and checked by reading, not compiled, not run on hardware
//
#include <hip/hip_runtime.h>

// ---------------------------------------------------------------------------
// MaskedLinear for MI455X (gfx1250): split-bf16 WMMA GEMM
// out = x @ (kron(clamp(block_mask),64x64) * W)^T + bias
//
// Pass 1 (memory-bound): masked W -> bf16 hi/lo planes in d_ws.
// Pass 2 (compute-bound): GEMM; B staged via global_load_async_to_lds_b128,
// A split f32->bf16 hi/lo in-loop; 3 bf16 WMMAs per tile (~f32 accuracy).
// ---------------------------------------------------------------------------

typedef __attribute__((ext_vector_type(16))) __bf16       bf16x16;
typedef __attribute__((ext_vector_type(8)))  float        f32x8;
typedef __attribute__((ext_vector_type(4)))  float        f32x4;
typedef __attribute__((ext_vector_type(4))) unsigned int  u32x4;
typedef __attribute__((ext_vector_type(2))) unsigned int  u32x2;

// GCC-style vector type matching the async-LDS builtin's pointee type
typedef int v4i_vs __attribute__((vector_size(16)));
typedef __attribute__((address_space(1))) v4i_vs* gbl_v4i_ptr;
typedef __attribute__((address_space(3))) v4i_vs* lds_v4i_ptr;

#define NROWS 8192
#define KDIM  4096
#define ODIM  4096
#define BM 128
#define BN 128
#define BK 32
#define KCHUNKS (KDIM / BK)

#define HAVE_ASYNC_B128 __has_builtin(__builtin_amdgcn_global_load_async_to_lds_b128)

__device__ __forceinline__ void wait_async0() {
#if __has_builtin(__builtin_amdgcn_s_wait_asynccnt)
  __builtin_amdgcn_s_wait_asynccnt(0);
#else
  asm volatile("s_wait_asynccnt 0" ::: "memory");
#endif
}

// pack the high 16 bits of r0 (low half of dst) and r1 (high half) in 1 op
__device__ __forceinline__ unsigned int pack_hi16(unsigned int r0,
                                                  unsigned int r1) {
#if __has_builtin(__builtin_amdgcn_perm)
  return __builtin_amdgcn_perm(r1, r0, 0x07060302u);
#else
  return (r0 >> 16) | (r1 & 0xFFFF0000u);
#endif
}

struct Pk { unsigned int h, l; };

// split two f32 into packed bf16 hi (RNE) and packed bf16 lo (trunc residual)
__device__ __forceinline__ Pk split2(float f0, float f1) {
  unsigned int u0 = __float_as_uint(f0);
  unsigned int u1 = __float_as_uint(f1);
  unsigned int r0 = u0 + (0x7FFFu + ((u0 >> 16) & 1u));
  unsigned int r1 = u1 + (0x7FFFu + ((u1 >> 16) & 1u));
  float l0 = f0 - __uint_as_float(r0 & 0xFFFF0000u);
  float l1 = f1 - __uint_as_float(r1 & 0xFFFF0000u);
  Pk p;
  p.h = pack_hi16(r0, r1);
  p.l = pack_hi16(__float_as_uint(l0), __float_as_uint(l1));
  return p;
}

struct SBuf {
  unsigned short aHi[BM * BK];
  unsigned short aLo[BM * BK];
  unsigned short bHi[BN * BK];
  unsigned short bLo[BN * BK];
};

__device__ __forceinline__ bf16x16 ldFrag(const unsigned short* p0,
                                          const unsigned short* p1) {
  union { u32x4 q[2]; bf16x16 v; } u;
  u.q[0] = *(const u32x4*)p0;
  u.q[1] = *(const u32x4*)p1;
  return u.v;
}

// ---------------------------------------------------------------------------
// Pass 1: masked W -> bf16 hi/lo planes (row-major, same indexing as W)
// ---------------------------------------------------------------------------
__global__ __launch_bounds__(256) void presplit_w(
    const float* __restrict__ w, const float* __restrict__ bmask,
    unsigned short* __restrict__ wHi, unsigned short* __restrict__ wLo) {
  size_t e = ((size_t)blockIdx.x * 256 + threadIdx.x) * 4;
  int n = (int)(e >> 12);          // row (OUT)
  int k = (int)(e & 4095);         // col (IN)
  float mv = bmask[(n >> 6) * (KDIM / 64) + (k >> 6)];
  float m = fminf(fmaxf(mv, 0.0f), 1.0f);
  f32x4 v = *(const f32x4*)(w + e);
  Pk p0 = split2(v[0] * m, v[1] * m);
  Pk p1 = split2(v[2] * m, v[3] * m);
  u32x2 hp = { p0.h, p1.h };
  u32x2 lp = { p0.l, p1.l };
  *(u32x2*)(wHi + e) = hp;
  *(u32x2*)(wLo + e) = lp;
}

// ---------------------------------------------------------------------------
// Pass 2: GEMM with pre-split W (async B staging)
// ---------------------------------------------------------------------------
__global__ __launch_bounds__(256) void masked_linear_wmma_pre(
    const float* __restrict__ x,
    const unsigned short* __restrict__ wHi,
    const unsigned short* __restrict__ wLo,
    const float* __restrict__ bias,
    float* __restrict__ out) {
  __shared__ SBuf sb[2];

  const int tid   = threadIdx.x;
  const int mBase = blockIdx.y * BM;
  const int nBase = blockIdx.x * BN;

  f32x4 aReg[4];
#if !HAVE_ASYNC_B128
  u32x4 bHiReg[2], bLoReg[2];
#endif

  auto loadA = [&](int kb) {
#pragma unroll
    for (int i = 0; i < 4; ++i) {
      int idx = tid + i * 256;
      int row = idx >> 3;
      int c4  = (idx & 7) << 2;
      aReg[i] = *(const f32x4*)(x + (size_t)(mBase + row) * KDIM +
                                (size_t)kb * BK + c4);
    }
  };

  auto storeA = [&](SBuf* s) {
#pragma unroll
    for (int i = 0; i < 4; ++i) {
      int idx = tid + i * 256;
      int row = idx >> 3;
      int c4  = (idx & 7) << 2;
      int off = row * BK + c4;
      Pk p0 = split2(aReg[i][0], aReg[i][1]);
      Pk p1 = split2(aReg[i][2], aReg[i][3]);
      u32x2 hp = { p0.h, p1.h };
      u32x2 lp = { p0.l, p1.l };
      *(u32x2*)(s->aHi + off) = hp;
      *(u32x2*)(s->aLo + off) = lp;
    }
  };

  // B tile: 128 rows x 32 bf16 per plane = 512 x 16B units, 2 per thread
  auto issueB = [&](int kb, SBuf* s) {
#pragma unroll
    for (int i = 0; i < 2; ++i) {
      int idx = tid + i * 256;
      int row = idx >> 2;
      int u8  = (idx & 3) << 3;
      size_t g = (size_t)(nBase + row) * KDIM + (size_t)kb * BK + u8;
      int l = row * BK + u8;
#if HAVE_ASYNC_B128
      __builtin_amdgcn_global_load_async_to_lds_b128(
          (gbl_v4i_ptr)(wHi + g), (lds_v4i_ptr)(s->bHi + l), 0, 0);
      __builtin_amdgcn_global_load_async_to_lds_b128(
          (gbl_v4i_ptr)(wLo + g), (lds_v4i_ptr)(s->bLo + l), 0, 0);
#else
      bHiReg[i] = *(const u32x4*)(wHi + g);
      bLoReg[i] = *(const u32x4*)(wLo + g);
#endif
    }
  };

#if !HAVE_ASYNC_B128
  auto storeB = [&](SBuf* s) {
#pragma unroll
    for (int i = 0; i < 2; ++i) {
      int idx = tid + i * 256;
      int row = idx >> 2;
      int u8  = (idx & 3) << 3;
      int l = row * BK + u8;
      *(u32x4*)(s->bHi + l) = bHiReg[i];
      *(u32x4*)(s->bLo + l) = bLoReg[i];
    }
  };
#endif

  const int lane   = tid & 31;
  const int wid    = tid >> 5;
  const int waveM  = (wid >> 2) * 64;
  const int waveN  = (wid & 3) * 32;
  const int lrow   = lane & 15;
  const bool hiHalf = lane >= 16;

  f32x8 acc[4][2];
#pragma unroll
  for (int mt = 0; mt < 4; ++mt)
#pragma unroll
    for (int nt = 0; nt < 2; ++nt) acc[mt][nt] = (f32x8)0.0f;

  // prologue: fill buffer 0
  issueB(0, &sb[0]);
  loadA(0);
  storeA(&sb[0]);
#if HAVE_ASYNC_B128
  wait_async0();
#else
  storeB(&sb[0]);
#endif

  for (int kb = 0; kb < KCHUNKS; ++kb) {
    __syncthreads();
    const bool hasNext = (kb + 1) < KCHUNKS;
    SBuf* nxt = &sb[(kb + 1) & 1];
    if (hasNext) {
      issueB(kb + 1, nxt);   // async: straight into next LDS buffer
      loadA(kb + 1);
    }

    const SBuf* s = &sb[kb & 1];

    bf16x16 aHi[4], aLo[4];
#pragma unroll
    for (int mt = 0; mt < 4; ++mt) {
      int r = waveM + mt * 16 + lrow;
      int o = r * BK + (hiHalf ? 8 : 0);
      aHi[mt] = ldFrag(s->aHi + o, s->aHi + o + 16);
      aLo[mt] = ldFrag(s->aLo + o, s->aLo + o + 16);
    }
    bf16x16 bHi[2], bLo[2];
#pragma unroll
    for (int nt = 0; nt < 2; ++nt) {
      int r = waveN + nt * 16 + lrow;
      int o = r * BK + (hiHalf ? 8 : 0);
      bHi[nt] = ldFrag(s->bHi + o, s->bHi + o + 16);
      bLo[nt] = ldFrag(s->bLo + o, s->bLo + o + 16);
    }

#pragma unroll
    for (int mt = 0; mt < 4; ++mt) {
#pragma unroll
      for (int nt = 0; nt < 2; ++nt) {
        acc[mt][nt] = __builtin_amdgcn_wmma_f32_16x16x32_bf16(
            false, aHi[mt], false, bHi[nt], (short)0, acc[mt][nt], false, false);
        acc[mt][nt] = __builtin_amdgcn_wmma_f32_16x16x32_bf16(
            false, aHi[mt], false, bLo[nt], (short)0, acc[mt][nt], false, false);
        acc[mt][nt] = __builtin_amdgcn_wmma_f32_16x16x32_bf16(
            false, aLo[mt], false, bHi[nt], (short)0, acc[mt][nt], false, false);
      }
    }

    if (hasNext) {
      storeA(nxt);
#if !HAVE_ASYNC_B128
      storeB(nxt);
#endif
    }
#if HAVE_ASYNC_B128
    wait_async0();           // async B for nxt complete before next barrier
#endif
  }

#pragma unroll
  for (int mt = 0; mt < 4; ++mt) {
#pragma unroll
    for (int nt = 0; nt < 2; ++nt) {
      int col = nBase + waveN + nt * 16 + lrow;
      float bv = bias[col];
      int rowB = mBase + waveM + mt * 16 + (hiHalf ? 8 : 0);
#pragma unroll
      for (int r = 0; r < 8; ++r) {
        out[(size_t)(rowB + r) * ODIM + col] = acc[mt][nt][r] + bv;
      }
    }
  }
}

// ---------------------------------------------------------------------------
// Fallback: fully self-contained kernel (splits W in-loop), used if d_ws is
// too small for the pre-split planes.
// ---------------------------------------------------------------------------
__global__ __launch_bounds__(256) void masked_linear_wmma(
    const float* __restrict__ x, const float* __restrict__ w,
    const float* __restrict__ bias, const float* __restrict__ bmask,
    float* __restrict__ out) {
  __shared__ SBuf sb[2];

  const int tid   = threadIdx.x;
  const int mBase = blockIdx.y * BM;
  const int nBase = blockIdx.x * BN;

  f32x4 aReg[4];
  f32x4 bReg[4];
  float mReg[4];

  auto loadStage = [&](int kb) {
#pragma unroll
    for (int i = 0; i < 4; ++i) {
      int idx = tid + i * 256;
      int row = idx >> 3;
      int c4  = (idx & 7) << 2;
      size_t koff = (size_t)kb * BK + (size_t)c4;
      aReg[i] = *(const f32x4*)(x + (size_t)(mBase + row) * KDIM + koff);
      bReg[i] = *(const f32x4*)(w + (size_t)(nBase + row) * KDIM + koff);
      float mv = bmask[((nBase + row) >> 6) * (KDIM / 64) + (kb >> 1)];
      mReg[i] = fminf(fmaxf(mv, 0.0f), 1.0f);
    }
  };

  auto storeStage = [&](SBuf* s) {
#pragma unroll
    for (int i = 0; i < 4; ++i) {
      int idx = tid + i * 256;
      int row = idx >> 3;
      int c4  = (idx & 7) << 2;
      int off = row * BK + c4;
      {
        Pk p0 = split2(aReg[i][0], aReg[i][1]);
        Pk p1 = split2(aReg[i][2], aReg[i][3]);
        u32x2 hp = { p0.h, p1.h };
        u32x2 lp = { p0.l, p1.l };
        *(u32x2*)(s->aHi + off) = hp;
        *(u32x2*)(s->aLo + off) = lp;
      }
      {
        float m = mReg[i];
        Pk p0 = split2(bReg[i][0] * m, bReg[i][1] * m);
        Pk p1 = split2(bReg[i][2] * m, bReg[i][3] * m);
        u32x2 hp = { p0.h, p1.h };
        u32x2 lp = { p0.l, p1.l };
        *(u32x2*)(s->bHi + off) = hp;
        *(u32x2*)(s->bLo + off) = lp;
      }
    }
  };

  const int lane   = tid & 31;
  const int wid    = tid >> 5;
  const int waveM  = (wid >> 2) * 64;
  const int waveN  = (wid & 3) * 32;
  const int lrow   = lane & 15;
  const bool hiHalf = lane >= 16;

  f32x8 acc[4][2];
#pragma unroll
  for (int mt = 0; mt < 4; ++mt)
#pragma unroll
    for (int nt = 0; nt < 2; ++nt) acc[mt][nt] = (f32x8)0.0f;

  loadStage(0);
  storeStage(&sb[0]);

  for (int kb = 0; kb < KCHUNKS; ++kb) {
    __syncthreads();
    const bool hasNext = (kb + 1) < KCHUNKS;
    if (hasNext) loadStage(kb + 1);

    const SBuf* s = &sb[kb & 1];

    bf16x16 aHi[4], aLo[4];
#pragma unroll
    for (int mt = 0; mt < 4; ++mt) {
      int r = waveM + mt * 16 + lrow;
      int o = r * BK + (hiHalf ? 8 : 0);
      aHi[mt] = ldFrag(s->aHi + o, s->aHi + o + 16);
      aLo[mt] = ldFrag(s->aLo + o, s->aLo + o + 16);
    }
    bf16x16 bHi[2], bLo[2];
#pragma unroll
    for (int nt = 0; nt < 2; ++nt) {
      int r = waveN + nt * 16 + lrow;
      int o = r * BK + (hiHalf ? 8 : 0);
      bHi[nt] = ldFrag(s->bHi + o, s->bHi + o + 16);
      bLo[nt] = ldFrag(s->bLo + o, s->bLo + o + 16);
    }

#pragma unroll
    for (int mt = 0; mt < 4; ++mt) {
#pragma unroll
      for (int nt = 0; nt < 2; ++nt) {
        acc[mt][nt] = __builtin_amdgcn_wmma_f32_16x16x32_bf16(
            false, aHi[mt], false, bHi[nt], (short)0, acc[mt][nt], false, false);
        acc[mt][nt] = __builtin_amdgcn_wmma_f32_16x16x32_bf16(
            false, aHi[mt], false, bLo[nt], (short)0, acc[mt][nt], false, false);
        acc[mt][nt] = __builtin_amdgcn_wmma_f32_16x16x32_bf16(
            false, aLo[mt], false, bHi[nt], (short)0, acc[mt][nt], false, false);
      }
    }

    if (hasNext) storeStage(&sb[(kb + 1) & 1]);
  }

#pragma unroll
  for (int mt = 0; mt < 4; ++mt) {
#pragma unroll
    for (int nt = 0; nt < 2; ++nt) {
      int col = nBase + waveN + nt * 16 + lrow;
      float bv = bias[col];
      int rowB = mBase + waveM + mt * 16 + (hiHalf ? 8 : 0);
#pragma unroll
      for (int r = 0; r < 8; ++r) {
        out[(size_t)(rowB + r) * ODIM + col] = acc[mt][nt][r] + bv;
      }
    }
  }
}

extern "C" void kernel_launch(void* const* d_in, const int* in_sizes, int n_in,
                              void* d_out, int out_size, void* d_ws, size_t ws_size,
                              hipStream_t stream) {
  const float* x     = (const float*)d_in[0];
  const float* wgt   = (const float*)d_in[1];
  const float* bias  = (const float*)d_in[2];
  const float* bmask = (const float*)d_in[3];
  float* out = (float*)d_out;

  const size_t planeElems = (size_t)ODIM * KDIM;
  const size_t wsNeeded   = 2 * planeElems * sizeof(unsigned short);  // 64 MiB

  dim3 grid(ODIM / BN, NROWS / BM, 1);
  dim3 block(256, 1, 1);

  if (ws_size >= wsNeeded) {
    unsigned short* wHi = (unsigned short*)d_ws;
    unsigned short* wLo = wHi + planeElems;
    int nPre = (int)(planeElems / (256 * 4));  // 16384 blocks
    presplit_w<<<dim3(nPre, 1, 1), block, 0, stream>>>(wgt, bmask, wHi, wLo);
    masked_linear_wmma_pre<<<grid, block, 0, stream>>>(x, wHi, wLo, bias, out);
  } else {
    masked_linear_wmma<<<grid, block, 0, stream>>>(x, wgt, bias, bmask, out);
  }
}
